// GAU_43954695308013
// MI455X (gfx1250) — compile-verified, matrix operand order checked by reading, not run on hardware
//
#include <hip/hip_runtime.h>
#include <hip/hip_bf16.h>
#include <math.h>

#define B_    20
#define S_    500
#define DIM_  300
#define HALF_ 150
#define QK_   128
#define HID_  600
#define BS_   (B_ * S_)
#define KD_   320
#define SP_   512
#define KH_   640
#define NW_   608
#define NO_   304

typedef __attribute__((ext_vector_type(16))) _Float16 v16h;
typedef __attribute__((ext_vector_type(8)))  float    v8f;
typedef __attribute__((ext_vector_type(8)))  int      v8i;
typedef __attribute__((ext_vector_type(16))) int      v16i;

__constant__ unsigned char FP8TAB[16] = {
  0x00, 0x38, 0x40, 0x44, 0x48, 0x4A, 0x4C, 0x4E,
  0x50, 0x51, 0x52, 0x53, 0x54, 0x55, 0x56, 0x57
};

__device__ inline float block_sum(float v, float* buf) {
  int t = threadIdx.x, n = blockDim.x;
  buf[t] = v; __syncthreads();
  for (int s2 = n >> 1; s2 > 0; s2 >>= 1) {
    if (t < s2) buf[t] += buf[t + s2];
    __syncthreads();
  }
  float r = buf[0]; __syncthreads();
  return r;
}
__device__ inline float block_max(float v, float* buf) {
  int t = threadIdx.x, n = blockDim.x;
  buf[t] = v; __syncthreads();
  for (int s2 = n >> 1; s2 > 0; s2 >>= 1) {
    if (t < s2) buf[t] = fmaxf(buf[t], buf[t + s2]);
    __syncthreads();
  }
  float r = buf[0]; __syncthreads();
  return r;
}

__device__ inline v16h load_f16_frag(const _Float16* base, int ld, int row, int k0, int hi) {
  const _Float16* p = base + (size_t)row * ld + k0 + hi * 8;
  union U { uint4 u; _Float16 h[8]; } lo, hw;
  lo.u = *(const uint4*)p;
  hw.u = *(const uint4*)(p + 16);
  v16h r;
#pragma unroll
  for (int i = 0; i < 8; ++i) { r[i] = lo.h[i]; r[8 + i] = hw.h[i]; }
  return r;
}
__device__ inline v8i load_b8_frag64(const unsigned char* base, int ld, int row, int k0, int hi) {
  const unsigned char* p = base + (size_t)row * ld + k0 + hi * 8;
  int2 c0 = *(const int2*)(p);
  int2 c1 = *(const int2*)(p + 16);
  int2 c2 = *(const int2*)(p + 32);
  int2 c3 = *(const int2*)(p + 48);
  v8i r;
  r[0] = c0.x; r[1] = c0.y; r[2] = c1.x; r[3] = c1.y;
  r[4] = c2.x; r[5] = c2.y; r[6] = c3.x; r[7] = c3.y;
  return r;
}
__device__ inline v16i load_fp8_frag128(const unsigned char* base, int ld, int row, int k0, int hi) {
  v8i lo = load_b8_frag64(base, ld, row, k0, hi);
  v8i up = load_b8_frag64(base, ld, row, k0 + 64, hi);
  v16i r;
#pragma unroll
  for (int i = 0; i < 8; ++i) { r[i] = lo[i]; r[8 + i] = up[i]; }
  return r;
}

__device__ inline float silu_f(float z) { return z / (1.f + __expf(-z)); }
__device__ inline float quant15(float y) { return truncf(fminf(fmaxf(y, -15.f), 15.f)); }
__device__ inline unsigned char enc_fp8(float y) {
  int v = (int)y;
  unsigned s = (v < 0) ? 0x80u : 0u;
  int m = (v < 0) ? -v : v;
  return (unsigned char)(s | FP8TAB[m]);
}

__device__ inline float t5_bias(int i, int j, const float* __restrict__ rel_emb) {
  int n = i - j;
  int ret = (n < 0) ? 16 : 0;
  int an = (n < 0) ? -n : n;
  int bucket;
  if (an < 8) bucket = ret + an;
  else {
    int vl = 8 + (int)(__logf((float)an * 0.125f) * (8.f / __logf(16.f)));
    if (vl > 15) vl = 15;
    bucket = ret + vl;
  }
  return rel_emb[bucket] * 11.313708498984761f;
}

__global__ void k_convert_pad(const float* __restrict__ src, _Float16* __restrict__ dst,
                              int rows, int cols, int rowsPad, int colsPad) {
  int idx = blockIdx.x * blockDim.x + threadIdx.x;
  int total = rowsPad * colsPad;
  if (idx >= total) return;
  int r = idx / colsPad, c = idx % colsPad;
  float v = (r < rows && c < cols) ? src[(size_t)r * cols + c] : 0.f;
  dst[idx] = (_Float16)v;
}

__global__ void __launch_bounds__(128) k_ln_shift_quant(
    const float* __restrict__ x, const float* __restrict__ gamma, const float* __restrict__ beta,
    const float* __restrict__ qk_s, const float* __restrict__ hid_s,
    _Float16* __restrict__ xq16, _Float16* __restrict__ xh16,
    float* __restrict__ sq, float* __restrict__ sh) {
  __shared__ float red[128];
  __shared__ float shifted[DIM_];
  int m = blockIdx.x;
  int s = m % S_;
  const float* row  = x + (size_t)m * DIM_;
  const float* prev = row - DIM_;
  bool hasPrev = (s > 0);
  float a = 0.f, a2 = 0.f, p = 0.f, p2 = 0.f;
  for (int d = threadIdx.x; d < DIM_; d += 128) {
    float v = row[d]; a += v; a2 += v * v;
    if (hasPrev) { float w = prev[d]; p += w; p2 += w * w; }
  }
  float suma  = block_sum(a,  red);
  float suma2 = block_sum(a2, red);
  float sump  = block_sum(p,  red);
  float sump2 = block_sum(p2, red);
  float mu   = suma / DIM_;
  float var  = suma2 / DIM_ - mu * mu;
  float rinv = rsqrtf(var + 1e-5f);
  float pmu   = sump / DIM_;
  float pvar  = sump2 / DIM_ - pmu * pmu;
  float prinv = rsqrtf(pvar + 1e-5f);
  for (int d = threadIdx.x; d < DIM_; d += 128) {
    float v;
    if (d < HALF_) v = hasPrev ? ((prev[d] - pmu) * prinv * gamma[d] + beta[d]) : 0.f;
    else           v = (row[d] - mu) * rinv * gamma[d] + beta[d];
    shifted[d] = v;
  }
  __syncthreads();
  float mq = 0.f, mh = 0.f;
  for (int d = threadIdx.x; d < DIM_; d += 128) {
    mq = fmaxf(mq, fabsf(shifted[d] / qk_s[d]));
    mh = fmaxf(mh, fabsf(shifted[d] / hid_s[d]));
  }
  float s_q = block_max(mq, red) / 15.f;
  float s_h = block_max(mh, red) / 15.f;
  if (threadIdx.x == 0) { sq[m] = s_q; sh[m] = s_h; }
  float iq = s_q > 0.f ? 1.f / s_q : 0.f;
  float ih = s_h > 0.f ? 1.f / s_h : 0.f;
  for (int d = threadIdx.x; d < KD_; d += 128) {
    _Float16 vq = (_Float16)0.f, vh = (_Float16)0.f;
    if (d < DIM_) {
      vq = (_Float16)quant15(shifted[d] / qk_s[d]  * iq);
      vh = (_Float16)quant15(shifted[d] / hid_s[d] * ih);
    }
    xq16[(size_t)m * KD_ + d] = vq;
    xh16[(size_t)m * KD_ + d] = vh;
  }
}

__global__ void __launch_bounds__(256) k_gemm_h(
    const _Float16* __restrict__ A, const _Float16* __restrict__ W,
    const float* __restrict__ bh, const float* __restrict__ sh,
    const float* __restrict__ hparams,
    float* __restrict__ vbuf, float* __restrict__ gate) {
  const int NT = (2 * HID_) / 16;
  int wave = threadIdx.x >> 5, lane = threadIdx.x & 31;
  int tile = blockIdx.x * 8 + wave;
  if (tile >= (BS_ / 16) * NT) return;
  int tm = (tile / NT) * 16, tn = (tile % NT) * 16;
  int hi = lane >> 4, l15 = lane & 15;
  v8f acc = {};
  for (int k0 = 0; k0 < KD_; k0 += 32) {
    if (k0 + 32 < KD_)
      __builtin_prefetch(A + (size_t)(tm + l15) * KD_ + k0 + 32, 0, 1);
    v16h af = load_f16_frag(A, KD_, tm + l15, k0, hi);
    v16h bf = load_f16_frag(W, KD_, tn + l15, k0, hi);
    acc = __builtin_amdgcn_wmma_f32_16x16x32_f16(false, af, false, bf, (short)0, acc, false, false);
  }
  int n = tn + l15;
  float scn = hparams[n], bn = bh[n];
#pragma unroll
  for (int j = 0; j < 8; ++j) {
    int m = tm + hi * 8 + j;
    float val = silu_f(acc[j] + bn) * sh[m] * scn;
    if (n < HID_) vbuf[(size_t)m * HID_ + n] = val;
    else          gate[(size_t)m * HID_ + (n - HID_)] = val;
  }
}

__global__ void __launch_bounds__(256) k_gemm_qk(
    const _Float16* __restrict__ A, const _Float16* __restrict__ W,
    const float* __restrict__ bqk, const float* __restrict__ sq,
    const float* __restrict__ qparams, float* __restrict__ qkbuf) {
  const int NT = QK_ / 16;
  int wave = threadIdx.x >> 5, lane = threadIdx.x & 31;
  int tile = blockIdx.x * 8 + wave;
  if (tile >= (BS_ / 16) * NT) return;
  int tm = (tile / NT) * 16, tn = (tile % NT) * 16;
  int hi = lane >> 4, l15 = lane & 15;
  v8f acc = {};
  for (int k0 = 0; k0 < KD_; k0 += 32) {
    v16h af = load_f16_frag(A, KD_, tm + l15, k0, hi);
    v16h bf = load_f16_frag(W, KD_, tn + l15, k0, hi);
    acc = __builtin_amdgcn_wmma_f32_16x16x32_f16(false, af, false, bf, (short)0, acc, false, false);
  }
  int n = tn + l15;
  float scn = qparams[n], bn = bqk[n];
#pragma unroll
  for (int j = 0; j < 8; ++j) {
    int m = tm + hi * 8 + j;
    qkbuf[(size_t)m * QK_ + n] = silu_f(acc[j] + bn) * sq[m] * scn;
  }
}

__global__ void __launch_bounds__(128) k_rotary_quant(
    const float* __restrict__ qkbuf,
    const float* __restrict__ os_gamma, const float* __restrict__ os_beta,
    unsigned char* __restrict__ qq8, unsigned char* __restrict__ kq8,
    float* __restrict__ qs, float* __restrict__ ks) {
  __shared__ float red[128];
  __shared__ float qsh[QK_], ksh[QK_];
  int idx = blockIdx.x;
  int b = idx / SP_, s = idx % SP_;
  int d = threadIdx.x;
  size_t orow = (size_t)idx * QK_;
  if (s >= S_) {
    qq8[orow + d] = 0; kq8[orow + d] = 0;
    if (d == 0) { qs[idx] = 0.f; ks[idx] = 0.f; }
    return;
  }
  float t = qkbuf[((size_t)(b * S_ + s)) * QK_ + d];
  qsh[d] = t * os_gamma[d]       + os_beta[d];
  ksh[d] = t * os_gamma[QK_ + d] + os_beta[QK_ + d];
  __syncthreads();
  float qv = qsh[d], kv = ksh[d];
  if (d < 32) {
    int tp = d >> 1;
    float invf = __powf(10000.f, -(float)(2 * tp) / 32.f);
    float ang = (float)s * invf;
    float c = __cosf(ang), si = __sinf(ang);
    if ((d & 1) == 0) { qv = qsh[d] * c - qsh[d + 1] * si; kv = ksh[d] * c - ksh[d + 1] * si; }
    else              { qv = qsh[d] * c + qsh[d - 1] * si; kv = ksh[d] * c + ksh[d - 1] * si; }
  }
  float s_q = block_max(fabsf(qv), red) / 15.f;
  float s_k = block_max(fabsf(kv), red) / 15.f;
  if (d == 0) { qs[idx] = s_q; ks[idx] = s_k; }
  float iq = s_q > 0.f ? 1.f / s_q : 0.f;
  float ik = s_k > 0.f ? 1.f / s_k : 0.f;
  qq8[orow + d] = enc_fp8(quant15(qv * iq));
  kq8[orow + d] = enc_fp8(quant15(kv * ik));
}

__global__ void __launch_bounds__(256) k_sim_attn(
    const unsigned char* __restrict__ qq8, const unsigned char* __restrict__ kq8,
    const float* __restrict__ qs, const float* __restrict__ ks,
    const float* __restrict__ rel_emb, float* __restrict__ attn) {
  const int NT = SP_ / 16;
  int wave = threadIdx.x >> 5, lane = threadIdx.x & 31;
  int tile = blockIdx.x * 8 + wave;
  if (tile >= B_ * NT * NT) return;
  int b = tile / (NT * NT);
  int r = tile % (NT * NT);
  int ti = (r / NT) * 16, tj = (r % NT) * 16;
  int hi = lane >> 4, l15 = lane & 15;
  const unsigned char* Ab = qq8 + (size_t)b * SP_ * QK_;
  const unsigned char* Bb = kq8 + (size_t)b * SP_ * QK_;
  v16i af = load_fp8_frag128(Ab, QK_, ti + l15, 0, hi);
  v16i bf = load_fp8_frag128(Bb, QK_, tj + l15, 0, hi);
  v8f acc = {};
  acc = __builtin_amdgcn_wmma_f32_16x16x128_fp8_fp8(af, bf, (short)0, acc, false, false);
  int j = tj + l15;
  float ksj = ks[b * SP_ + j];
#pragma unroll
  for (int jj = 0; jj < 8; ++jj) {
    int i = ti + hi * 8 + jj;
    float simf = acc[jj] * qs[b * SP_ + i] * ksj + t5_bias(i, j, rel_emb);
    float rl = fmaxf(simf * (1.f / (float)S_), 0.f);
    attn[((size_t)(b * SP_ + i)) * SP_ + j] = rl * rl;
  }
}

__global__ void __launch_bounds__(256) k_attn_quant(
    const float* __restrict__ attn, unsigned char* __restrict__ aq8, float* __restrict__ as_) {
  __shared__ float red[256];
  int idx = blockIdx.x;
  int i = idx % SP_;
  size_t row = (size_t)idx * SP_;
  if (i >= S_) {
    for (int j = threadIdx.x; j < SP_; j += 256) aq8[row + j] = 0;
    if (threadIdx.x == 0) as_[idx] = 0.f;
    return;
  }
  float mx = 0.f;
  for (int j = threadIdx.x; j < S_; j += 256) mx = fmaxf(mx, fabsf(attn[row + j]));
  float sc = block_max(mx, red) / 15.f;
  if (threadIdx.x == 0) as_[idx] = sc;
  float inv = sc > 0.f ? 1.f / sc : 0.f;
  for (int j = threadIdx.x; j < SP_; j += 256) {
    unsigned char q = 0;
    if (j < S_) q = enc_fp8(quant15(attn[row + j] * inv));
    aq8[row + j] = q;
  }
}

__global__ void __launch_bounds__(128) k_vquant(
    const float* __restrict__ vbuf, unsigned char* __restrict__ vqT, float* __restrict__ vs) {
  __shared__ float red[128];
  int idx = blockIdx.x;
  int b = idx / NW_, d = idx % NW_;
  size_t orow = (size_t)idx * SP_;
  if (d >= HID_) {
    for (int s = threadIdx.x; s < SP_; s += 128) vqT[orow + s] = 0;
    if (threadIdx.x == 0) vs[idx] = 0.f;
    return;
  }
  float mx = 0.f;
  for (int s = threadIdx.x; s < S_; s += 128)
    mx = fmaxf(mx, fabsf(vbuf[((size_t)(b * S_ + s)) * HID_ + d]));
  float sc = block_max(mx, red) / 15.f;
  if (threadIdx.x == 0) vs[idx] = sc;
  float inv = sc > 0.f ? 1.f / sc : 0.f;
  for (int s = threadIdx.x; s < SP_; s += 128) {
    unsigned char q = 0;
    if (s < S_) q = enc_fp8(quant15(vbuf[((size_t)(b * S_ + s)) * HID_ + d] * inv));
    vqT[orow + s] = q;
  }
}

__global__ void __launch_bounds__(256) k_out_gemm(
    const unsigned char* __restrict__ aq8, const unsigned char* __restrict__ vqT,
    const float* __restrict__ as_, const float* __restrict__ vs,
    const float* __restrict__ gate, const float* __restrict__ out_s,
    float* __restrict__ outbuf) {
  const int NT = NW_ / 16;
  const int MT = SP_ / 16;
  int wave = threadIdx.x >> 5, lane = threadIdx.x & 31;
  int tile = blockIdx.x * 8 + wave;
  if (tile >= B_ * MT * NT) return;
  int b = tile / (MT * NT);
  int r = tile % (MT * NT);
  int ti = (r / NT) * 16, tn = (r % NT) * 16;
  int hi = lane >> 4, l15 = lane & 15;
  const unsigned char* Ab = aq8 + (size_t)b * SP_ * SP_;
  const unsigned char* Bb = vqT + (size_t)b * NW_ * SP_;
  v8f acc = {};
  for (int k0 = 0; k0 < SP_; k0 += 128) {
    if (k0 + 128 < SP_)
      __builtin_prefetch(Ab + (size_t)(ti + l15) * SP_ + k0 + 128, 0, 1);
    v16i af = load_fp8_frag128(Ab, SP_, ti + l15, k0, hi);
    v16i bf = load_fp8_frag128(Bb, SP_, tn + l15, k0, hi);
    acc = __builtin_amdgcn_wmma_f32_16x16x128_fp8_fp8(af, bf, (short)0, acc, false, false);
  }
  int n = tn + l15;
  if (n >= HID_) return;
  float vsn = vs[b * NW_ + n];
  float oin = 1.f / out_s[n];
#pragma unroll
  for (int jj = 0; jj < 8; ++jj) {
    int i = ti + hi * 8 + jj;
    if (i < S_) {
      float val = acc[jj] * as_[b * SP_ + i] * vsn;
      val = val * gate[((size_t)(b * S_ + i)) * HID_ + n] * oin;
      outbuf[((size_t)(b * SP_ + i)) * HID_ + n] = val;
    }
  }
}

__global__ void __launch_bounds__(256) k_out_quant(
    const float* __restrict__ outbuf, _Float16* __restrict__ oq16, float* __restrict__ osc) {
  __shared__ float red[256];
  int m = blockIdx.x;
  int b = m / S_, s = m % S_;
  size_t irow = ((size_t)(b * SP_ + s)) * HID_;
  float mx = 0.f;
  for (int n = threadIdx.x; n < HID_; n += 256) mx = fmaxf(mx, fabsf(outbuf[irow + n]));
  float sc = block_max(mx, red) / 15.f;
  if (threadIdx.x == 0) osc[m] = sc;
  float inv = sc > 0.f ? 1.f / sc : 0.f;
  for (int n = threadIdx.x; n < KH_; n += 256) {
    _Float16 q = (_Float16)0.f;
    if (n < HID_) q = (_Float16)quant15(outbuf[irow + n] * inv);
    oq16[(size_t)m * KH_ + n] = q;
  }
}

__global__ void __launch_bounds__(256) k_final_gemm(
    const _Float16* __restrict__ oq16, const _Float16* __restrict__ Wo16,
    const float* __restrict__ bo, const float* __restrict__ osc,
    const float* __restrict__ out_params, const float* __restrict__ x,
    float* __restrict__ dout) {
  const int NT = NO_ / 16;
  int wave = threadIdx.x >> 5, lane = threadIdx.x & 31;
  int tile = blockIdx.x * 8 + wave;
  if (tile >= (BS_ / 16) * NT) return;
  int tm = (tile / NT) * 16, tn = (tile % NT) * 16;
  int hi = lane >> 4, l15 = lane & 15;
  v8f acc = {};
  for (int k0 = 0; k0 < KH_; k0 += 32) {
    if (k0 + 32 < KH_)
      __builtin_prefetch(oq16 + (size_t)(tm + l15) * KH_ + k0 + 32, 0, 1);
    v16h af = load_f16_frag(oq16, KH_, tm + l15, k0, hi);
    v16h bf = load_f16_frag(Wo16, KH_, tn + l15, k0, hi);
    acc = __builtin_amdgcn_wmma_f32_16x16x32_f16(false, af, false, bf, (short)0, acc, false, false);
  }
  int n = tn + l15;
  if (n >= DIM_) return;
  float bon = bo[n], opn = out_params[n];
#pragma unroll
  for (int j = 0; j < 8; ++j) {
    int m = tm + hi * 8 + j;
    dout[(size_t)m * DIM_ + n] = (acc[j] + bon) * osc[m] * opn + x[(size_t)m * DIM_ + n];
  }
}

extern "C" void kernel_launch(void* const* d_in, const int* in_sizes, int n_in,
                              void* d_out, int out_size, void* d_ws, size_t ws_size,
                              hipStream_t stream) {
  (void)in_sizes; (void)n_in; (void)out_size; (void)ws_size;
  const float* x        = (const float*)d_in[0];
  const float* ngamma   = (const float*)d_in[1];
  const float* nbeta    = (const float*)d_in[2];
  const float* Wh       = (const float*)d_in[3];
  const float* bh       = (const float*)d_in[4];
  const float* Wqk      = (const float*)d_in[5];
  const float* bqk      = (const float*)d_in[6];
  const float* os_gamma = (const float*)d_in[7];
  const float* os_beta  = (const float*)d_in[8];
  const float* Wo       = (const float*)d_in[9];
  const float* bo       = (const float*)d_in[10];
  const float* rel_emb  = (const float*)d_in[11];
  const float* qk_s     = (const float*)d_in[12];
  const float* hidden_s = (const float*)d_in[13];
  const float* out_s    = (const float*)d_in[14];
  const float* hparams  = (const float*)d_in[15];
  const float* qparams  = (const float*)d_in[16];
  const float* oparams  = (const float*)d_in[17];
  float* dout = (float*)d_out;

  char* base = (char*)d_ws;
  size_t off = 0;
  auto alloc = [&](size_t bytes) -> void* {
    off = (off + 255) & ~(size_t)255;
    void* p = base + off;
    off += bytes;
    return p;
  };

  _Float16* Wh16  = (_Float16*)alloc((size_t)2 * HID_ * KD_ * sizeof(_Float16));
  _Float16* Wqk16 = (_Float16*)alloc((size_t)QK_ * KD_ * sizeof(_Float16));
  _Float16* Wo16  = (_Float16*)alloc((size_t)NO_ * KH_ * sizeof(_Float16));
  _Float16* xq16  = (_Float16*)alloc((size_t)BS_ * KD_ * sizeof(_Float16));
  _Float16* xh16  = (_Float16*)alloc((size_t)BS_ * KD_ * sizeof(_Float16));
  float* sq       = (float*)alloc((size_t)BS_ * sizeof(float));
  float* sh       = (float*)alloc((size_t)BS_ * sizeof(float));
  float* vbuf     = (float*)alloc((size_t)BS_ * HID_ * sizeof(float));
  float* gate     = (float*)alloc((size_t)BS_ * HID_ * sizeof(float));
  float* qkbuf    = (float*)alloc((size_t)BS_ * QK_ * sizeof(float));
  unsigned char* qq8 = (unsigned char*)alloc((size_t)B_ * SP_ * QK_);
  unsigned char* kq8 = (unsigned char*)alloc((size_t)B_ * SP_ * QK_);
  float* qs       = (float*)alloc((size_t)B_ * SP_ * sizeof(float));
  float* ks       = (float*)alloc((size_t)B_ * SP_ * sizeof(float));
  float* attnbuf  = (float*)alloc((size_t)B_ * SP_ * SP_ * sizeof(float));
  unsigned char* aq8 = (unsigned char*)alloc((size_t)B_ * SP_ * SP_);
  float* as_      = (float*)alloc((size_t)B_ * SP_ * sizeof(float));
  unsigned char* vqT = (unsigned char*)alloc((size_t)B_ * NW_ * SP_);
  float* vs       = (float*)alloc((size_t)B_ * NW_ * sizeof(float));
  float* outbuf   = (float*)alloc((size_t)B_ * SP_ * HID_ * sizeof(float));
  _Float16* oq16  = (_Float16*)alloc((size_t)BS_ * KH_ * sizeof(_Float16));
  float* osc      = (float*)alloc((size_t)BS_ * sizeof(float));

  {
    int t0 = 2 * HID_ * KD_;
    k_convert_pad<<<(t0 + 255) / 256, 256, 0, stream>>>(Wh, Wh16, 2 * HID_, DIM_, 2 * HID_, KD_);
    int t1 = QK_ * KD_;
    k_convert_pad<<<(t1 + 255) / 256, 256, 0, stream>>>(Wqk, Wqk16, QK_, DIM_, QK_, KD_);
    int t2 = NO_ * KH_;
    k_convert_pad<<<(t2 + 255) / 256, 256, 0, stream>>>(Wo, Wo16, DIM_, HID_, NO_, KH_);
  }
  k_ln_shift_quant<<<BS_, 128, 0, stream>>>(x, ngamma, nbeta, qk_s, hidden_s, xq16, xh16, sq, sh);
  k_gemm_h<<<((BS_ / 16) * ((2 * HID_) / 16) + 7) / 8, 256, 0, stream>>>(
      xh16, Wh16, bh, sh, hparams, vbuf, gate);
  k_gemm_qk<<<((BS_ / 16) * (QK_ / 16) + 7) / 8, 256, 0, stream>>>(
      xq16, Wqk16, bqk, sq, qparams, qkbuf);
  k_rotary_quant<<<B_ * SP_, 128, 0, stream>>>(qkbuf, os_gamma, os_beta, qq8, kq8, qs, ks);
  k_sim_attn<<<(B_ * (SP_ / 16) * (SP_ / 16) + 7) / 8, 256, 0, stream>>>(
      qq8, kq8, qs, ks, rel_emb, attnbuf);
  k_attn_quant<<<B_ * SP_, 256, 0, stream>>>(attnbuf, aq8, as_);
  k_vquant<<<B_ * NW_, 128, 0, stream>>>(vbuf, vqT, vs);
  k_out_gemm<<<(B_ * (SP_ / 16) * (NW_ / 16) + 7) / 8, 256, 0, stream>>>(
      aq8, vqT, as_, vs, gate, out_s, outbuf);
  k_out_quant<<<BS_, 256, 0, stream>>>(outbuf, oq16, osc);
  k_final_gemm<<<((BS_ / 16) * (NO_ / 16) + 7) / 8, 256, 0, stream>>>(
      oq16, Wo16, bo, osc, oparams, x, dout);
}
